// QuantizedLinear_22522808500941
// MI455X (gfx1250) — compile-verified, matrix operand order checked by reading, not run on hardware
//
#include <hip/hip_runtime.h>
#include <hip/hip_bf16.h>

typedef __attribute__((ext_vector_type(16))) _Float16 v16h;
typedef __attribute__((ext_vector_type(8)))  _Float16 v8h;
typedef __attribute__((ext_vector_type(8)))  float    v8f;
typedef __attribute__((ext_vector_type(4)))  unsigned int u32x4;
typedef __attribute__((ext_vector_type(8)))  int          i32x8;
typedef __attribute__((ext_vector_type(4)))  int          i32x4;

constexpr int ROWS = 8192;   // Bz*S
constexpr int N    = 4096;   // input features
constexpr int M    = 4096;   // output features
constexpr int RANK = 64;

#if defined(__has_builtin)
#if __has_builtin(__builtin_amdgcn_tensor_load_to_lds) && \
    __has_builtin(__builtin_amdgcn_s_wait_tensorcnt)
#define USE_TDM 1
#endif
#endif
#ifndef USE_TDM
#define USE_TDM 0
#endif

// ---------------------------------------------------------------------------
// K1: x' = fwht(x * SU) / 64  -> f16   (one block per row, FWHT in LDS)
// ---------------------------------------------------------------------------
__global__ __launch_bounds__(256)
void hada_in_kernel(const float* __restrict__ in, const float* __restrict__ SU,
                    _Float16* __restrict__ xh)
{
    __shared__ float buf[N];
    const int row = blockIdx.x;
    const int tid = threadIdx.x;
    const float* src = in + (size_t)row * N;
    for (int i = tid; i < N; i += 256) buf[i] = src[i] * SU[i];
    __syncthreads();
    for (int h = 1; h < N; h <<= 1) {
        for (int p = tid; p < N / 2; p += 256) {
            int i = ((p & ~(h - 1)) << 1) | (p & (h - 1));
            int j = i + h;
            float a = buf[i], b = buf[j];
            buf[i] = a + b;
            buf[j] = a - b;
        }
        __syncthreads();
    }
    _Float16* dst = xh + (size_t)row * N;
    for (int i = tid; i < N; i += 256)
        dst[i] = (_Float16)(buf[i] * 0.015625f);   // 1/sqrt(4096)
}

// ---------------------------------------------------------------------------
// K2: Weff[m][k] = Wscale * codebook[Qidxs[m][k/8]][k%8] + (A@B)[m][k] -> f16
// ---------------------------------------------------------------------------
constexpr int DM = 32;
constexpr int DK = 128;

__global__ __launch_bounds__(256)
void decode_weff_kernel(const int* __restrict__ Qidxs,
                        const float* __restrict__ cb,
                        const float* __restrict__ A,
                        const float* __restrict__ Bmat,
                        const float* __restrict__ wscale,
                        _Float16* __restrict__ Weff)
{
    __shared__ float sA[DM][RANK];   // 8 KB
    __shared__ float sB[RANK][DK];   // 32 KB
    const int m0 = blockIdx.y * DM;
    const int k0 = blockIdx.x * DK;
    const int tid = threadIdx.x;

    for (int i = tid; i < DM * RANK; i += 256)
        sA[i / RANK][i % RANK] = A[(size_t)(m0 + i / RANK) * RANK + (i % RANK)];
    for (int i = tid; i < RANK * DK; i += 256)
        sB[i / DK][i % DK] = Bmat[(size_t)(i / DK) * N + k0 + (i % DK)];
    __syncthreads();

    const float ws = wscale[0];
    const int ki = tid % DK;
    const int mg = tid / DK;
    for (int mi = mg * 16; mi < mg * 16 + 16; ++mi) {
        float acc = 0.f;
#pragma unroll
        for (int r = 0; r < RANK; ++r) acc += sA[mi][r] * sB[r][ki];
        const int m = m0 + mi;
        const int k = k0 + ki;
        const int idx = Qidxs[(size_t)m * (N / 8) + (k >> 3)];
        const float w = cb[idx * 8 + (k & 7)];
        Weff[(size_t)m * N + k] = (_Float16)(ws * w + acc);
    }
}

// ---------------------------------------------------------------------------
// K3: Z = X' @ Weff^T via v_wmma_f32_16x16x32_f16.
//     Tiles staged into LDS by the Tensor Data Mover (double-buffered,
//     TENSORcnt-pipelined); hardware pad reproduces the 144 B LDS row stride.
// ---------------------------------------------------------------------------
constexpr int BM = 128;
constexpr int BN = 128;
constexpr int BK = 64;
constexpr int LDT = BK + 8;   // 72 f16 = 144 B padded row stride

union alignas(32) Frag {
    v16h v;
    v8h  h[2];
};

#if USE_TDM
// Issue one TDM descriptor: 2-D tile of 128 rows x 64 f16 (128 B/row) from a
// row-major f16 tensor with row stride N elements, into LDS at lds_off, with
// 16 B of padding inserted after every 128 B stored (pad_interval=4 -> 32
// DWORDs, pad_amount=3 -> 4 DWORDs), matching LDT = 72 f16.
__device__ __forceinline__ void tdm_load_tile(const _Float16* gptr,
                                              unsigned lds_off,
                                              unsigned tensor_rows)
{
    const unsigned long long ga = (unsigned long long)(uintptr_t)gptr;
    u32x4 g0;
    g0[0] = 1u;                                    // count=1, user descriptor
    g0[1] = lds_off;                               // LDS byte address
    g0[2] = (unsigned)(ga & 0xFFFFFFFFu);          // global_addr[31:0]
    g0[3] = (unsigned)((ga >> 32) & 0x01FFFFFFu)   // global_addr[56:32]
          | (2u << 30);                            // type = 2 (image)
    i32x8 g1;
    g1[0] = (int)((1u << 16)        // data_size = 1 -> 2-byte elements
                | (1u << 20)        // pad_enable
                | (4u << 22)        // pad_interval: 32 DWORDs (128 B)
                | (3u << 25));      // pad_amount:   4 DWORDs (16 B)
    g1[1] = (int)(((unsigned)N & 0xFFFFu) << 16);          // tensor_dim0 lo16
    g1[2] = (int)(((unsigned)N >> 16)                      // tensor_dim0 hi16
                | ((tensor_rows & 0xFFFFu) << 16));        // tensor_dim1 lo16
    g1[3] = (int)((tensor_rows >> 16)                      // tensor_dim1 hi16
                | (64u << 16));                            // tile_dim0 = 64
    g1[4] = 128;                                           // tile_dim1 = 128
    g1[5] = N;                                             // tensor_dim0_stride
    g1[6] = 0;
    g1[7] = 0;
    i32x4 gz = {0, 0, 0, 0};
#if __clang_major__ >= 23
    i32x8 gz8 = {0, 0, 0, 0, 0, 0, 0, 0};
    __builtin_amdgcn_tensor_load_to_lds(g0, g1, gz, gz, gz8, 0);
#else
    __builtin_amdgcn_tensor_load_to_lds(g0, g1, gz, gz, 0);
#endif
}
#endif

__device__ __forceinline__ void copy_tile_fallback(_Float16* dst,
                                                   const _Float16* src,
                                                   int tid)
{
#pragma unroll
    for (int j = 0; j < 4; ++j) {
        const int c   = tid + j * 256;
        const int row = c >> 3;
        const int cc  = c & 7;
        *(float4*)&dst[row * LDT + cc * 8] =
            *(const float4*)&src[(size_t)row * N + cc * 8];
    }
}

__global__ __launch_bounds__(256)
void gemm_wmma_kernel(const _Float16* __restrict__ X,
                      const _Float16* __restrict__ W,
                      float* __restrict__ Z)
{
    __shared__ _Float16 sX[2][BM * LDT];   // 2 x 18 KB
    __shared__ _Float16 sW[2][BN * LDT];   // 2 x 18 KB

    const int tid  = threadIdx.x;
    const int lane = tid & 31;
    const int wave = tid >> 5;      // 0..7
    const int wm   = wave & 3;      // 4 waves along M, 32 rows each
    const int wn   = wave >> 2;     // 2 waves along N, 64 cols each
    const int block_m = blockIdx.y * BM;
    const int block_n = blockIdx.x * BN;

    const _Float16* Xbase = X + (size_t)block_m * N;
    const _Float16* Wbase = W + (size_t)block_n * N;

    // 16-bit operand lane layout (ISA 7.12.2): row = lane&15,
    // K chunks at kb and kb+16 where kb = (lane&16) ? 8 : 0.
    const int frag_row = lane & 15;
    const int frag_kb  = (lane & 16) ? 8 : 0;

    v8f acc[2][4] = {};

#if USE_TDM
    const unsigned ldsX[2] = { (unsigned)(uintptr_t)&sX[0][0],
                               (unsigned)(uintptr_t)&sX[1][0] };
    const unsigned ldsW[2] = { (unsigned)(uintptr_t)&sW[0][0],
                               (unsigned)(uintptr_t)&sW[1][0] };
    if (wave == 0) {                       // TDM ignores EXEC; one wave issues
        tdm_load_tile(Xbase, ldsX[0], (unsigned)ROWS);
        tdm_load_tile(Wbase, ldsW[0], (unsigned)M);
    }
#else
    copy_tile_fallback(&sX[0][0], Xbase, tid);
    copy_tile_fallback(&sW[0][0], Wbase, tid);
#endif

    int cur = 0;
    for (int k0 = 0; k0 < N; k0 += BK) {
        const bool has_next = (k0 + BK) < N;
#if USE_TDM
        if (wave == 0) {
            if (has_next) {
                tdm_load_tile(Xbase + k0 + BK, ldsX[cur ^ 1], (unsigned)ROWS);
                tdm_load_tile(Wbase + k0 + BK, ldsW[cur ^ 1], (unsigned)M);
                // 2 ops just issued may remain outstanding; anything older
                // (the current buffer's pair) must have completed.
                __builtin_amdgcn_s_wait_tensorcnt(2);
            } else {
                __builtin_amdgcn_s_wait_tensorcnt(0);
            }
        }
#else
        if (has_next) {
            copy_tile_fallback(&sX[cur ^ 1][0], Xbase + k0 + BK, tid);
            copy_tile_fallback(&sW[cur ^ 1][0], Wbase + k0 + BK, tid);
        }
#endif
        __syncthreads();   // current buffer ready for all waves

        const _Float16* bx = &sX[cur][0];
        const _Float16* bw = &sW[cur][0];

        Frag fa[2][2];   // [m-tile][k-chunk]
        Frag fb[4][2];   // [n-tile][k-chunk]
#pragma unroll
        for (int ti = 0; ti < 2; ++ti)
#pragma unroll
            for (int kc = 0; kc < 2; ++kc) {
                const _Float16* p =
                    &bx[(wm * 32 + ti * 16 + frag_row) * LDT + kc * 32 + frag_kb];
                fa[ti][kc].h[0] = *(const v8h*)p;
                fa[ti][kc].h[1] = *(const v8h*)(p + 16);
            }
#pragma unroll
        for (int ni = 0; ni < 4; ++ni)
#pragma unroll
            for (int kc = 0; kc < 2; ++kc) {
                const _Float16* p =
                    &bw[(wn * 64 + ni * 16 + frag_row) * LDT + kc * 32 + frag_kb];
                fb[ni][kc].h[0] = *(const v8h*)p;
                fb[ni][kc].h[1] = *(const v8h*)(p + 16);
            }

#pragma unroll
        for (int kc = 0; kc < 2; ++kc)
#pragma unroll
            for (int ti = 0; ti < 2; ++ti)
#pragma unroll
                for (int ni = 0; ni < 4; ++ni)
                    acc[ti][ni] = __builtin_amdgcn_wmma_f32_16x16x32_f16(
                        false, fa[ti][kc].v, false, fb[ni][kc].v,
                        (short)0, acc[ti][ni], false, false);

        __syncthreads();   // all reads of `cur` done before it is re-staged
        cur ^= 1;
    }

    // C/D layout: VGPR i holds M = i (+8 for lanes 16-31), N = lane&15.
    const int mhalf = (lane & 16) ? 8 : 0;
#pragma unroll
    for (int ti = 0; ti < 2; ++ti)
#pragma unroll
        for (int ni = 0; ni < 4; ++ni) {
            const int col = block_n + wn * 64 + ni * 16 + (lane & 15);
#pragma unroll
            for (int i = 0; i < 8; ++i) {
                const int row = block_m + wm * 32 + ti * 16 + mhalf + i;
                Z[(size_t)row * M + col] = acc[ti][ni][i];
            }
        }
}

// ---------------------------------------------------------------------------
// K4: z = fwht(z) / 64 * SV   (in place, one block per row)
// ---------------------------------------------------------------------------
__global__ __launch_bounds__(256)
void hada_out_kernel(float* __restrict__ z, const float* __restrict__ SV)
{
    __shared__ float buf[M];
    const int row = blockIdx.x;
    const int tid = threadIdx.x;
    float* ptr = z + (size_t)row * M;
    for (int i = tid; i < M; i += 256) buf[i] = ptr[i];
    __syncthreads();
    for (int h = 1; h < M; h <<= 1) {
        for (int p = tid; p < M / 2; p += 256) {
            int i = ((p & ~(h - 1)) << 1) | (p & (h - 1));
            int j = i + h;
            float a = buf[i], b = buf[j];
            buf[i] = a + b;
            buf[j] = a - b;
        }
        __syncthreads();
    }
    for (int i = tid; i < M; i += 256)
        ptr[i] = buf[i] * 0.015625f * SV[i];
}

// ---------------------------------------------------------------------------
extern "C" void kernel_launch(void* const* d_in, const int* in_sizes, int n_in,
                              void* d_out, int out_size, void* d_ws, size_t ws_size,
                              hipStream_t stream)
{
    const float* input  = (const float*)d_in[0];   // (4,2048,4096)
    const int*   Qidxs  = (const int*)  d_in[1];   // (4096,512)
    const float* cb     = (const float*)d_in[2];   // (256,8)
    const float* SU     = (const float*)d_in[3];   // (4096,)
    const float* SV     = (const float*)d_in[4];   // (4096,)
    const float* Wscale = (const float*)d_in[5];   // scalar
    const float* A      = (const float*)d_in[6];   // (4096,64)
    const float* B      = (const float*)d_in[7];   // (64,4096)
    float* Z = (float*)d_out;                      // (8192,4096)

    _Float16* xh = (_Float16*)d_ws;
    _Float16* wh = (_Float16*)((char*)d_ws + (size_t)ROWS * N * sizeof(_Float16));

    hada_in_kernel<<<ROWS, 256, 0, stream>>>(input, SU, xh);
    decode_weff_kernel<<<dim3(N / DK, M / DM), 256, 0, stream>>>(
        Qidxs, cb, A, B, Wscale, wh);
    gemm_wmma_kernel<<<dim3(N / BN, ROWS / BM), 256, 0, stream>>>(xh, wh, Z);
    hada_out_kernel<<<ROWS, 256, 0, stream>>>(Z, SV);
}
